// Pyramid_44547400794764
// MI455X (gfx1250) — matrix operand order repaired and checked by reading, not compiled, not run on hardware
//
#include <hip/hip_runtime.h>
#include <hip/hip_bf16.h>

typedef _Float16 half_t;
typedef __attribute__((ext_vector_type(8)))  _Float16 v8h;
typedef __attribute__((ext_vector_type(16))) _Float16 v16h;
typedef __attribute__((ext_vector_type(8)))  float    v8f;

#define BM 128          // block tile M (4 waves x 32)
#define BN 32           // block tile N
#define TK 64           // K depth staged per iteration

// ---------------------------------------------------------------------------
// Implicit-GEMM 3D conv, SAME padding, optional stride on D.
// Compile-time CIN / K3 so all tap/cin decompositions are shifts & constants.
// Block: 128 threads = 4 wave32s stacked along M; each wave computes a
// 32Mx32N register tile (2x2 accumulators) -> 4 WMMAs per 32-k sub-chunk,
// 2 ds_load_b128 per WMMA of fragment traffic.
// X  : [B, Di, Hi, Wi, CIN]  f16 channels-last
// Wt : [Cout][Kpad]          f16, k = tap*CIN + cin, tap = (dz*K3+dy)*K3+dx
// Msk: [B, Do, Ho, Wo]       f16 (output-res mask) or null
// Sc : shortcut (f16 channels-last, output shape) or null
// Yh : f16 channels-last out; Yf: f32 NCDHW out (used if non-null)
// flags: bit0 = relu on input gather, bit1 = relu on output
// ---------------------------------------------------------------------------
template <int CIN, int K3>
__global__ __launch_bounds__(128)
void conv3d_wmma(const half_t* __restrict__ X, const half_t* __restrict__ Wt,
                 const half_t* __restrict__ Msk, const half_t* __restrict__ Sc,
                 half_t* __restrict__ Yh, float* __restrict__ Yf,
                 int Bsz, int Di, int Hi, int Wi,
                 int Do, int Ho, int Wo, int Cout,
                 int sd, int Kpad, int flags)
{
    constexpr int KEFF = K3 * K3 * K3 * CIN;
    constexpr int PADW = (K3 - 1) / 2;
    constexpr int TT   = K3 * K3;

    __shared__ __align__(32) half_t lA[BM * TK];   // 16 KB
    __shared__ __align__(32) half_t lB[BN * TK];   //  4 KB

    const int M      = Bsz * Do * Ho * Wo;
    const int m_base = blockIdx.x * BM;
    const int n_base = blockIdx.y * BN;
    const int tid    = threadIdx.x;
    const int lane   = tid & 31;           // wave32
    const int wave   = tid >> 5;           // 0..3, rows [wave*32, wave*32+32)
    const bool relu_in = (flags & 1) != 0;

    // A staging: one row per thread
    const int gm = m_base + tid;
    const bool mvalid = gm < M;
    int ob = 0, od = 0, oh = 0, ow = 0;
    {
        int t = mvalid ? gm : 0;
        ow = t % Wo; t /= Wo;
        oh = t % Ho; t /= Ho;
        od = t % Do; ob = t / Do;
    }
    const int id0 = od * sd - PADW, ih0 = oh - PADW, iw0 = ow - PADW;

    v8f acc[2][2] = {};

    for (int kb = 0; kb < Kpad; kb += TK) {
        // ---------------- stage A (128 rows x 64 k) ----------------
        #pragma unroll
        for (int aq = 0; aq < 4; ++aq) {
            v16h av = {};
            const int kt = kb + aq * 16;
            if constexpr (CIN >= 16) {
                // a 16-run of k never crosses a tap boundary (CIN % 16 == 0)
                if (mvalid && kt < KEFF) {
                    const int tap = kt / CIN;          // compile-time divisor
                    const int ci0 = kt % CIN;
                    const int dz = tap / TT;
                    const int rm = tap % TT;
                    const int dy = rm / K3, dx = rm % K3;
                    const int id = id0 + dz, ih = ih0 + dy, iw = iw0 + dx;
                    if ((unsigned)id < (unsigned)Di && (unsigned)ih < (unsigned)Hi &&
                        (unsigned)iw < (unsigned)Wi) {
                        const half_t* p = X +
                            (((((long)ob * Di + id) * Hi + ih) * Wi + iw) * (long)CIN + ci0);
                        av = *(const v16h*)p;
                    }
                }
            } else {
                // CIN == 1: tap == k, all decompositions constant-fold
                if (mvalid) {
                    #pragma unroll
                    for (int e = 0; e < 16; ++e) {
                        const int k = kt + e;
                        half_t v = (half_t)0.f;
                        if (k < KEFF) {
                            const int dz = k / TT;
                            const int rm = k % TT;
                            const int dy = rm / K3, dx = rm % K3;
                            const int id = id0 + dz, ih = ih0 + dy, iw = iw0 + dx;
                            if ((unsigned)id < (unsigned)Di && (unsigned)ih < (unsigned)Hi &&
                                (unsigned)iw < (unsigned)Wi)
                                v = X[(((long)ob * Di + id) * Hi + ih) * (long)Wi + iw];
                        }
                        av[e] = v;
                    }
                }
            }
            if (relu_in) {
                #pragma unroll
                for (int e = 0; e < 16; ++e)
                    av[e] = av[e] > (half_t)0.f ? av[e] : (half_t)0.f;
            }
            *(v16h*)&lA[tid * TK + aq * 16] = av;
        }
        // ---------------- stage B (32 cols x 64 k) ----------------
        {
            const int bn = tid >> 2, bq = tid & 3;
            v16h bv = {};
            const int gn = n_base + bn;
            if (gn < Cout)
                bv = *(const v16h*)(Wt + (long)gn * Kpad + kb + bq * 16);
            *(v16h*)&lB[bn * TK + bq * 16] = bv;
        }
        __syncthreads();
        // ------------- 2 sub-chunks x (2x2) WMMA 16x16x32 -------------
        {
            const int hi  = (lane >> 4) & 1;
            const int lnl = lane & 15;
            #pragma unroll
            for (int s = 0; s < 2; ++s) {
                v16h a[2], b[2];
                #pragma unroll
                for (int i = 0; i < 2; ++i) {
                    // A frag: lanes 0-15 row M, K[0..15];
                    // lanes 16-31 same rows, K[16..31]
                    const half_t* ar = &lA[(wave * 32 + i * 16 + lnl) * TK + s * 32];
                    a[i] = *(const v16h*)(ar + hi * 16);
                }
                #pragma unroll
                for (int j = 0; j < 2; ++j)
                    // B frag: lane = column N, 16 consecutive K per lane
                    b[j] = *(const v16h*)(&lB[(j * 16 + lnl) * TK + s * 32 + hi * 16]);
                #pragma unroll
                for (int i = 0; i < 2; ++i)
                    #pragma unroll
                    for (int j = 0; j < 2; ++j)
                        acc[i][j] = __builtin_amdgcn_wmma_f32_16x16x32_f16(
                            false, a[i], false, b[j], (short)0, acc[i][j], false, false);
            }
        }
        __syncthreads();
    }

    // ---------------- epilogue: *mask, +shortcut, relu, store ----------------
    {
        const int nl = lane & 15;
        const int mh = (lane >> 4) * 8;    // C/D layout: VGPR q -> M = q (+8 hi lanes)
        #pragma unroll
        for (int i = 0; i < 2; ++i) {
            #pragma unroll
            for (int j = 0; j < 2; ++j) {
                const int gn = n_base + j * 16 + nl;
                if (gn < Cout) {
                    #pragma unroll
                    for (int q = 0; q < 8; ++q) {
                        const int gm2 = m_base + wave * 32 + i * 16 + mh + q;
                        if (gm2 < M) {
                            float v = acc[i][j][q];
                            if (Msk) v *= (float)Msk[gm2];
                            if (Sc)  v += (float)Sc[(long)gm2 * Cout + gn];
                            if (flags & 2) v = v > 0.f ? v : 0.f;
                            if (Yf) {
                                int t = gm2;
                                int w2 = t % Wo; t /= Wo;
                                int h2 = t % Ho; t /= Ho;
                                int d2 = t % Do; int b2 = t / Do;
                                Yf[((((long)b2 * Cout + gn) * Do + d2) * Ho + h2) * Wo + w2] = v;
                            } else {
                                Yh[(long)gm2 * Cout + gn] = (half_t)v;
                            }
                        }
                    }
                }
            }
        }
    }
}

// ---------------------------------------------------------------------------
// Elementwise / setup kernels
// ---------------------------------------------------------------------------
__global__ __launch_bounds__(256)
void zero_k(half_t* p, long n)
{
    long i = (long)blockIdx.x * blockDim.x + threadIdx.x;
    if (i < n) p[i] = (half_t)0.f;
}

__global__ __launch_bounds__(256)
void scatter_k(const float* __restrict__ feats, const int* __restrict__ coors,
               half_t* __restrict__ grid, half_t* __restrict__ mask,
               int V, int Dd, int Hh, int Ww)
{
    int i = blockIdx.x * blockDim.x + threadIdx.x;
    if (i >= V) return;
    int b = coors[i * 4 + 0], z = coors[i * 4 + 1];
    int y = coors[i * 4 + 2], x = coors[i * 4 + 3];
    long idx = (((long)b * Dd + z) * Hh + y) * Ww + x;
    grid[idx] = (half_t)feats[i];   // coords are a permutation -> unique
    mask[idx] = (half_t)1.f;
}

// fused: relu -> maxpool(2,2,2) -> mask maxpool -> multiply by pooled mask
__global__ __launch_bounds__(256)
void pool_k(const half_t* __restrict__ in, const half_t* __restrict__ mi,
            half_t* __restrict__ out, half_t* __restrict__ mo,
            int Bsz, int Do, int Ho, int Wo, int C)
{
    long total = (long)Bsz * Do * Ho * Wo * C;
    long i = (long)blockIdx.x * blockDim.x + threadIdx.x;
    if (i >= total) return;
    int c = (int)(i % C); long t = i / C;
    int w = (int)(t % Wo); t /= Wo;
    int h = (int)(t % Ho); t /= Ho;
    int d = (int)(t % Do); int b = (int)(t / Do);
    const int Di = 2 * Do, Hi = 2 * Ho, Wi = 2 * Wo;
    float m = 0.f, mk = 0.f;
    for (int dz = 0; dz < 2; ++dz)
        for (int dy = 0; dy < 2; ++dy)
            for (int dx = 0; dx < 2; ++dx) {
                long base = ((((long)b * Di + 2 * d + dz) * Hi + 2 * h + dy) * Wi + 2 * w + dx);
                float v = (float)in[base * C + c];
                v = v > 0.f ? v : 0.f;
                m = v > m ? v : m;
                float mv = (float)mi[base];
                mk = mv > mk ? mv : mk;
            }
    out[i] = (half_t)(m * mk);
    if (c == 0) mo[(((long)b * Do + d) * Ho + h) * Wo + w] = (half_t)mk;
}

// (O,I,k3,k3,k3) f32 -> [O][Kpad] f16, k = tap*I + ci
__global__ __launch_bounds__(256)
void wtransform_k(const float* __restrict__ w, half_t* __restrict__ wt,
                  int O, int I, int k3, int Kpad)
{
    int i = blockIdx.x * blockDim.x + threadIdx.x;
    int total = O * Kpad;
    if (i >= total) return;
    int o = i / Kpad, k = i % Kpad;
    int taps = k3 * k3 * k3;
    half_t v = (half_t)0.f;
    if (k < taps * I) {
        int tap = k / I, ci = k % I;
        v = (half_t)w[((long)o * I + ci) * taps + tap];
    }
    wt[i] = v;
}

// ---------------------------------------------------------------------------
// Host-side dispatch over compile-time (CIN, K3)
// ---------------------------------------------------------------------------
static void launch_conv(hipStream_t s, int Cin, int k3, dim3 g,
                        const half_t* X, const half_t* Wt, const half_t* Msk,
                        const half_t* Sc, half_t* Yh, float* Yf,
                        int B, int Di, int Hi, int Wi,
                        int Do, int Ho, int Wo, int Cout,
                        int sd, int Kpad, int flags)
{
#define CONV_CASE(C, K)                                                         \
    if (Cin == (C) && k3 == (K)) {                                              \
        conv3d_wmma<C, K><<<g, 128, 0, s>>>(X, Wt, Msk, Sc, Yh, Yf, B, Di, Hi,  \
                                            Wi, Do, Ho, Wo, Cout, sd, Kpad,     \
                                            flags);                             \
        return;                                                                 \
    }
    CONV_CASE(1, 1)   CONV_CASE(1, 3)
    CONV_CASE(16, 1)  CONV_CASE(16, 3)
    CONV_CASE(32, 1)  CONV_CASE(32, 3)
    CONV_CASE(64, 1)  CONV_CASE(64, 3)
    CONV_CASE(128, 3) CONV_CASE(224, 3) CONV_CASE(256, 3)
#undef CONV_CASE
}

extern "C" void kernel_launch(void* const* d_in, const int* in_sizes, int n_in,
                              void* d_out, int out_size, void* d_ws, size_t ws_size,
                              hipStream_t stream)
{
    (void)n_in; (void)out_size; (void)ws_size;
    const float* feats = (const float*)d_in[0];
    const int*   coors = (const int*)d_in[1];
    const int    V     = in_sizes[0];
    const int    B     = 2;

    // --- workspace carving (deterministic) ---
    char*  ws  = (char*)d_ws;
    size_t off = 0;
    auto alloc = [&](size_t bytes) -> void* {
        void* p = ws + off;
        off = (off + bytes + 255) & ~(size_t)255;
        return p;
    };
    const size_t ACT = (size_t)2 * 16 * 160 * 128 * 16;   // max activation halves
    const size_t MSK = (size_t)2 * 16 * 160 * 128;
    half_t* bufX  = (half_t*)alloc(ACT * 2);
    half_t* bufH  = (half_t*)alloc(ACT * 2);
    half_t* bufS  = (half_t*)alloc(ACT * 2);
    half_t* maskA = (half_t*)alloc(MSK * 2);
    half_t* maskB = (half_t*)alloc(MSK * 2);

    struct WI { int idx, O, I, k3; };
    const WI wis[23] = {
        { 3, 16,   1, 1}, { 4, 16,   1, 3}, { 5, 16,  16, 3}, { 6, 16,  16, 3}, { 7, 16,  16, 3},
        { 8, 32,  16, 1}, { 9, 32,  16, 3}, {10, 32,  32, 3}, {11, 32,  32, 3}, {12, 32,  32, 3},
        {13, 64,  32, 1}, {14, 64,  32, 3}, {15, 64,  64, 3}, {16, 64,  64, 3}, {17, 64,  64, 3},
        {18, 128, 64, 1}, {19, 128, 64, 3}, {20, 128, 128, 3}, {21, 128, 128, 3}, {22, 128, 128, 3},
        {23, 224, 128, 3}, {24, 256, 224, 3}, {25, 384, 256, 3}
    };
    half_t* wt[23]; int kpad[23];
    for (int i = 0; i < 23; ++i) {
        int keff = wis[i].k3 * wis[i].k3 * wis[i].k3 * wis[i].I;
        kpad[i]  = (keff + 63) & ~63;                 // TK = 64 granularity
        wt[i]    = (half_t*)alloc((size_t)wis[i].O * kpad[i] * 2);
        int total = wis[i].O * kpad[i];
        wtransform_k<<<(total + 255) / 256, 256, 0, stream>>>(
            (const float*)d_in[wis[i].idx], wt[i], wis[i].O, wis[i].I, wis[i].k3, kpad[i]);
    }

    // --- scatter into dense grid (C=1) + mask ---
    {
        long n = (long)B * 16 * 160 * 128;
        zero_k<<<(unsigned)((n + 255) / 256), 256, 0, stream>>>(bufX, n);
        zero_k<<<(unsigned)((n + 255) / 256), 256, 0, stream>>>(maskA, n);
        scatter_k<<<(V + 255) / 256, 256, 0, stream>>>(feats, coors, bufX, maskA, V, 16, 160, 128);
    }

    auto conv = [&](const half_t* Xp, int wi, const half_t* Mp, const half_t* Scp,
                    half_t* Yh, float* Yf, int Di, int Hi, int Wi, int Cin,
                    int Do, int sd, int flags) {
        int Cout = wis[wi].O, k3 = wis[wi].k3;
        long M = (long)B * Do * Hi * Wi;
        dim3 g((unsigned)((M + BM - 1) / BM), (unsigned)((Cout + BN - 1) / BN));
        launch_conv(stream, Cin, k3, g, Xp, wt[wi], Mp, Scp, Yh, Yf,
                    B, Di, Hi, Wi, Do, Hi, Wi, Cout, sd, kpad[wi], flags);
    };
    auto pool = [&](const half_t* in, const half_t* mi, half_t* outp, half_t* mo,
                    int Do, int Ho, int Wo, int C) {
        long total = (long)B * Do * Ho * Wo * C;
        pool_k<<<(unsigned)((total + 255) / 256), 256, 0, stream>>>(
            in, mi, outp, mo, B, Do, Ho, Wo, C);
    };

    int Dd = 16, Hh = 160, Ww = 128;
    // ---- stage 0: C 1 -> 16, mask = maskA ----
    conv(bufX, 0, nullptr, nullptr, bufS, nullptr, Dd, Hh, Ww, 1,  Dd, 1, 0); // nin shortcut
    conv(bufX, 1, maskA,   nullptr, bufH, nullptr, Dd, Hh, Ww, 1,  Dd, 1, 1); // w1(relu x)*m
    conv(bufH, 2, maskA,   bufS,    bufX, nullptr, Dd, Hh, Ww, 16, Dd, 1, 1); // w2(relu h)*m + S
    conv(bufX, 3, maskA,   nullptr, bufH, nullptr, Dd, Hh, Ww, 16, Dd, 1, 1); // rb1 w1
    conv(bufH, 4, maskA,   bufX,    bufS, nullptr, Dd, Hh, Ww, 16, Dd, 1, 1); // rb1 w2 + x
    pool(bufS, maskA, bufX, maskB, Dd / 2, Hh / 2, Ww / 2, 16);
    Dd /= 2; Hh /= 2; Ww /= 2;
    // ---- stage 1: C 16 -> 32, mask = maskB ----
    conv(bufX, 5, nullptr, nullptr, bufS, nullptr, Dd, Hh, Ww, 16, Dd, 1, 0);
    conv(bufX, 6, maskB,   nullptr, bufH, nullptr, Dd, Hh, Ww, 16, Dd, 1, 1);
    conv(bufH, 7, maskB,   bufS,    bufX, nullptr, Dd, Hh, Ww, 32, Dd, 1, 1);
    conv(bufX, 8, maskB,   nullptr, bufH, nullptr, Dd, Hh, Ww, 32, Dd, 1, 1);
    conv(bufH, 9, maskB,   bufX,    bufS, nullptr, Dd, Hh, Ww, 32, Dd, 1, 1);
    pool(bufS, maskB, bufX, maskA, Dd / 2, Hh / 2, Ww / 2, 32);
    Dd /= 2; Hh /= 2; Ww /= 2;
    // ---- stage 2: C 32 -> 64, mask = maskA ----
    conv(bufX, 10, nullptr, nullptr, bufS, nullptr, Dd, Hh, Ww, 32, Dd, 1, 0);
    conv(bufX, 11, maskA,   nullptr, bufH, nullptr, Dd, Hh, Ww, 32, Dd, 1, 1);
    conv(bufH, 12, maskA,   bufS,    bufX, nullptr, Dd, Hh, Ww, 64, Dd, 1, 1);
    conv(bufX, 13, maskA,   nullptr, bufH, nullptr, Dd, Hh, Ww, 64, Dd, 1, 1);
    conv(bufH, 14, maskA,   bufX,    bufS, nullptr, Dd, Hh, Ww, 64, Dd, 1, 1);
    pool(bufS, maskA, bufX, maskB, Dd / 2, Hh / 2, Ww / 2, 64);
    Dd /= 2; Hh /= 2; Ww /= 2;
    // ---- stage 3: C 64 -> 128, mask = maskB ----
    conv(bufX, 15, nullptr, nullptr, bufS, nullptr, Dd, Hh, Ww, 64,  Dd, 1, 0);
    conv(bufX, 16, maskB,   nullptr, bufH, nullptr, Dd, Hh, Ww, 64,  Dd, 1, 1);
    conv(bufH, 17, maskB,   bufS,    bufX, nullptr, Dd, Hh, Ww, 128, Dd, 1, 1);
    conv(bufX, 18, maskB,   nullptr, bufH, nullptr, Dd, Hh, Ww, 128, Dd, 1, 1);
    conv(bufH, 19, maskB,   bufX,    bufS, nullptr, Dd, Hh, Ww, 128, Dd, 1, 1);
    pool(bufS, maskB, bufX, maskA, Dd / 2, Hh / 2, Ww / 2, 128);
    Dd /= 2; Hh /= 2; Ww /= 2;            // -> (1, 10, 8)
    // ---- dense head: relu(conv) x3, stride (2,1,1) on D (D=1 -> D=1) ----
    conv(bufX, 20, nullptr, nullptr, bufH, nullptr, Dd, Hh, Ww, 128, 1, 2, 2);
    conv(bufH, 21, nullptr, nullptr, bufS, nullptr, Dd, Hh, Ww, 224, 1, 2, 2);
    conv(bufS, 22, nullptr, nullptr, nullptr, (float*)d_out, Dd, Hh, Ww, 256, 1, 2, 2);
}
